// DiffMultiHeadedAttention_56934086476580
// MI455X (gfx1250) — matrix operand order repaired and checked by reading, not compiled, not run on hardware
//
#include <hip/hip_runtime.h>
#include <cstdint>
#include <cstddef>

// ---------------------------------------------------------------------------
// CDNA5 (gfx1250) Differential Multi-Head Attention
//   bf16 WMMA (v_wmma_f32_16x16x32_bf16), f32 accumulation, wave32.
//   Big-LDS (~157KB) attention kernel, TDM tensor_load_to_lds for V tiles.
// ---------------------------------------------------------------------------

typedef __attribute__((ext_vector_type(16))) __bf16 v16bf;
typedef __attribute__((ext_vector_type(8)))  float  v8f;
typedef __attribute__((ext_vector_type(4)))  unsigned int u32x4;
typedef __attribute__((ext_vector_type(8)))  int i32x8;
typedef __attribute__((ext_vector_type(4)))  int i32x4;

#define WMMA_BF16(a, b, c) \
  __builtin_amdgcn_wmma_f32_16x16x32_bf16(false, (a), false, (b), (short)0, (c), false, false)

#if defined(__AMDGCN__) && __has_builtin(__builtin_amdgcn_tensor_load_to_lds) && \
    __has_builtin(__builtin_amdgcn_s_wait_tensorcnt)
#define USE_TDM 1
#else
#define USE_TDM 0
#endif

// ---- LDS fragment gathers (layouts per CDNA5 ISA) ---------------
// A 16x32 bf16: lane L (row M=L&15, half=L>>4) holds K in two 8-elem runs:
//   [8*half, +8) and [16+8*half, +8)  -> two ds_load_b128.
__device__ inline v16bf lds_frag_a(const __bf16* row, int half) {
  union { uint4 u[2]; v16bf v; } r;
  const char* p = (const char*)row + half * 16;
  r.u[0] = *(const uint4*)(p);
  r.u[1] = *(const uint4*)(p + 32);
  return r.v;
}
// B 32x16 bf16: lane L (col N=L&15, half=L>>4) holds K in two 8-elem runs:
//   [8*half, +8) and [16+8*half, +8)  -> two ds_load_b128.
__device__ inline v16bf lds_frag_b(const __bf16* row, int half) {
  union { uint4 u[2]; v16bf v; } r;
  const char* p = (const char*)row + half * 16;
  r.u[0] = *(const uint4*)(p);
  r.u[1] = *(const uint4*)(p + 32);
  return r.v;
}

__device__ inline v8f v8f_zero() {
  v8f z = {0.f, 0.f, 0.f, 0.f, 0.f, 0.f, 0.f, 0.f};
  return z;
}

// ---------------------------------------------------------------------------
// lambda = exp(sum(lq1*lk1)) - exp(sum(lq2*lk2)) + 0.8     (vectors len 64)
// ---------------------------------------------------------------------------
__global__ void lambda_kernel(const float* __restrict__ lq1, const float* __restrict__ lq2,
                              const float* __restrict__ lk1, const float* __restrict__ lk2,
                              float* __restrict__ lam) {
  int t = threadIdx.x; // one wave32
  float s1 = lq1[t] * lk1[t] + lq1[t + 32] * lk1[t + 32];
  float s2 = lq2[t] * lk2[t] + lq2[t + 32] * lk2[t + 32];
  for (int off = 16; off; off >>= 1) {
    s1 += __shfl_xor(s1, off, 32);
    s2 += __shfl_xor(s2, off, 32);
  }
  if (t == 0) *lam = __expf(s1) - __expf(s2) + 0.8f;
}

// ---------------------------------------------------------------------------
// bf16-WMMA GEMM:  Y[m][n] = sum_k A'[m][k] * W[n][k] + bias[n]
// Block tile 128x128, 8 waves in 4(M)x2(N), wave tile 32x64 (2x4 WMMA tiles).
// Register-pipelined staging: chunk k+1 global loads overlap chunk-k WMMAs.
// mode 0: A'=A (query), write Q split-head bf16 (1/sqrt(64) folded in)
// mode 1: A'=A (key),   write K split-head bf16
// mode 2: A'=A (value), write V transposed bf16  [b][hd][d(128)][n]
// mode 3: A = O-buffer; A'[m][c] = O[b][2hd][n][ck] - lam*O[b][2hd+1][n][ck]
//         (fused differential combine), write f32 `out`
// ---------------------------------------------------------------------------
#define GP 40   // LDS pitch (bf16): 32 + 8 pad

__global__ __launch_bounds__(256)
void gemm_bf16_kernel(const float* __restrict__ A, const float* __restrict__ W,
                      const float* __restrict__ bias, const float* __restrict__ lam,
                      int K, int mode,
                      __bf16* __restrict__ qs, __bf16* __restrict__ ks,
                      __bf16* __restrict__ vt, float* __restrict__ out) {
  __shared__ __bf16 a_lds[128 * GP];
  __shared__ __bf16 w_lds[128 * GP];

  const int tid  = threadIdx.x;
  const int wave = tid >> 5, lane = tid & 31;
  const int half = (lane >> 4) & 1, l16 = lane & 15;
  const int m0 = blockIdx.x * 128;
  const int n0 = blockIdx.y * 128;
  const int wm = wave & 3;   // M rows [wm*32, +32)
  const int wn = wave >> 2;  // N cols [wn*64, +64)
  const float lamv = (mode == 3) ? *lam : 0.0f;

  v8f acc[2][4];
#pragma unroll
  for (int mt = 0; mt < 2; ++mt)
#pragma unroll
    for (int nt = 0; nt < 4; ++nt) acc[mt][nt] = v8f_zero();

  float4 ra[4], rw[4];
  auto issue_loads = [&](int kc) {
#pragma unroll
    for (int j = 0; j < 4; ++j) {
      int qid  = tid + 256 * j;       // 0..1023
      int row  = qid >> 3;            // 0..127
      int quad = qid & 7;             // 4 floats each
      const float* wp = W + (size_t)(n0 + row) * K + kc + quad * 4;
      if (kc + 64 < K) __builtin_prefetch(wp + 64, 0, 1); // global_prefetch_b8
      rw[j] = *(const float4*)wp;
      if (mode == 3) {
        // fused differential combine while staging A
        int m = m0 + row, c = kc + quad * 4;
        int bb = m >> 10, n = m & 1023;
        int hd = c >> 7, ck = c & 127;
        size_t base = (((size_t)(bb * 16 + 2 * hd)) * 1024 + n) * 128 + ck;
        float4 o0 = *(const float4*)(A + base);
        float4 o1 = *(const float4*)(A + base + (size_t)128 * 1024);
        ra[j].x = o0.x - lamv * o1.x; ra[j].y = o0.y - lamv * o1.y;
        ra[j].z = o0.z - lamv * o1.z; ra[j].w = o0.w - lamv * o1.w;
      } else {
        const float* ap = A + (size_t)(m0 + row) * K + kc + quad * 4;
        if (kc + 64 < K) __builtin_prefetch(ap + 64, 0, 1);
        ra[j] = *(const float4*)ap;
      }
    }
  };

  issue_loads(0);
  for (int kc = 0; kc < K; kc += 32) {
    // ---- store staged registers -> LDS (bf16) ----
#pragma unroll
    for (int j = 0; j < 4; ++j) {
      int qid  = tid + 256 * j;
      int row  = qid >> 3;
      int quad = qid & 7;
      union { __bf16 h[4]; uint2 u; } ta, tw;
      ta.h[0] = (__bf16)ra[j].x; ta.h[1] = (__bf16)ra[j].y;
      ta.h[2] = (__bf16)ra[j].z; ta.h[3] = (__bf16)ra[j].w;
      tw.h[0] = (__bf16)rw[j].x; tw.h[1] = (__bf16)rw[j].y;
      tw.h[2] = (__bf16)rw[j].z; tw.h[3] = (__bf16)rw[j].w;
      *(uint2*)&a_lds[row * GP + quad * 4] = ta.u;
      *(uint2*)&w_lds[row * GP + quad * 4] = tw.u;
    }
    __syncthreads();

    // issue next chunk's global loads; latency hidden under the WMMA burst
    if (kc + 32 < K) issue_loads(kc + 32);

    // ---- preload all fragments, then 8 back-to-back WMMAs ----
    v16bf afrag[2], bfrag[4];
#pragma unroll
    for (int mt = 0; mt < 2; ++mt)
      afrag[mt] = lds_frag_a(&a_lds[(wm * 32 + mt * 16 + l16) * GP], half);
#pragma unroll
    for (int nt = 0; nt < 4; ++nt)
      bfrag[nt] = lds_frag_b(&w_lds[(wn * 64 + nt * 16 + l16) * GP], half);
#pragma unroll
    for (int nt = 0; nt < 4; ++nt)
#pragma unroll
      for (int mt = 0; mt < 2; ++mt)
        acc[mt][nt] = WMMA_BF16(afrag[mt], bfrag[nt], acc[mt][nt]);
    __syncthreads();
  }

  // ---- epilogue ----
#pragma unroll
  for (int mt = 0; mt < 2; ++mt) {
#pragma unroll
    for (int nt = 0; nt < 4; ++nt) {
      int gn = n0 + wn * 64 + nt * 16 + l16;
      float bv = bias[gn];
#pragma unroll
      for (int r = 0; r < 8; ++r) {
        int gm = m0 + wm * 32 + mt * 16 + r + 8 * half;
        float v = acc[mt][nt][r] + bv;
        if (mode == 3) {
          out[(size_t)gm * 1024 + gn] = v;
        } else {
          int b = gm >> 10, n = gm & 1023;
          int hd = gn >> 7, ck = gn & 127;
          if (mode == 2) {
            vt[(((size_t)(b * 8 + hd) * 128) + ck) * 1024 + n] = (__bf16)v;
          } else {
            int h2 = (hd << 1) | (ck >> 6), d = ck & 63;
            __bf16 o = (__bf16)(mode == 0 ? v * 0.125f : v);
            __bf16* dst = (mode == 0) ? qs : ks;
            dst[(((size_t)(b * 16 + h2) * 1024) + n) * 64 + d] = o;
          }
        }
      }
    }
  }
}

// ---------------------------------------------------------------------------
// Attention: per block, one (b, h2) sub-head and a 64-query stripe.
//   Phase 1: S = (scaled Q) K^T via WMMA -> bf16 stripe in LDS (64 x 1024)
//   Phase 2: row softmax in LDS
//   Phase 3: O = P @ V^T via WMMA (V tiles via TDM tensor_load_to_lds)
// LDS ~157 KB (CDNA5 320 KB WGP LDS). Dynamic segment assumed at LDS addr 0.
// ---------------------------------------------------------------------------
#define SP 1032  // score stripe pitch (bf16): 1024 + 8 pad
#define QP 72    // Q tile pitch: 64 + 8
#define KP 72    // K tile pitch: 64 + 8
#define VP 40    // V tile pitch: 32 + 8  (TDM pad: 16 DW interval + 4 DW pad)

#define S_BYTES  (64 * SP * 2)
#define Q_BYTES  (64 * QP * 2)
#define K_BYTES  (64 * KP * 2)
#define V_BYTES  (128 * VP * 2)
#define Q_OFF    (S_BYTES)
#define K_OFF    (S_BYTES + Q_BYTES)
#define V_OFF    (S_BYTES + Q_BYTES + K_BYTES)
#define ATTN_LDS (S_BYTES + Q_BYTES + K_BYTES + V_BYTES)   // 160,768 B

#if USE_TDM
// TDM D# for one V tile: 2D tile 32(x) x 128(y) of 2B elements out of a
// [128][1024] bf16 tensor, LDS padding 16B per 64B row -> pitch VP=40 bf16.
__device__ inline void tdm_load_v_tile(const __bf16* gsrc, unsigned lds_byte_off) {
  unsigned long long ga = (unsigned long long)(uintptr_t)gsrc;
  u32x4 g0;
  g0[0] = 1u;                                      // count=1, user descriptor
  g0[1] = lds_byte_off;                            // lds_addr
  g0[2] = (unsigned)(ga & 0xffffffffu);            // global_addr[31:0]
  g0[3] = (unsigned)((ga >> 32) & 0x01ffffffu)     // global_addr[56:32]
          | (2u << 30);                            // type=2 ("image")
  i32x8 g1;
  g1[0] = (1 << 16)            // data_size = 2 bytes
        | (1 << 20)            // pad_enable
        | (3 << 22)            // pad_interval = 16 DWORDs (64B row)
        | (3 << 25);           // pad_amount   = 4 DWORDs (16B)
  g1[1] = (int)(1024u << 16);  // tensor_dim0 = 1024 (low16 @ [63:48])
  g1[2] = (int)(128u << 16);   // dim0 hi=0 | tensor_dim1 = 128
  g1[3] = (int)(32u << 16);    // dim1 hi=0 | tile_dim0 = 32
  g1[4] = 128;                 // tile_dim1 = 128, tile_dim2 = 0
  g1[5] = 1024;                // tensor_dim0_stride[31:0] = 1024
  g1[6] = 0;                   // stride hi | dim1_stride lo
  g1[7] = 0;
  i32x4 z4 = {0, 0, 0, 0};
#if __clang_major__ >= 23
  i32x8 z8 = {0, 0, 0, 0, 0, 0, 0, 0};
  __builtin_amdgcn_tensor_load_to_lds(g0, g1, z4, z4, z8, 0);
#else
  __builtin_amdgcn_tensor_load_to_lds(g0, g1, z4, z4, 0);
#endif
}
#endif

__global__ __launch_bounds__(256)
void attn_kernel(const __bf16* __restrict__ qs, const __bf16* __restrict__ ks,
                 const __bf16* __restrict__ vt, float* __restrict__ obuf) {
  extern __shared__ char smem[];
  __bf16* s_lds = (__bf16*)smem;
  __bf16* q_lds = (__bf16*)(smem + Q_OFF);
  __bf16* k_lds = (__bf16*)(smem + K_OFF);
  __bf16* v_lds = (__bf16*)(smem + V_OFF);

  const int tid  = threadIdx.x;
  const int wave = tid >> 5, lane = tid & 31;
  const int half = lane >> 4, l16 = lane & 15;
  const int q0   = blockIdx.x * 64;
  const int bh2  = blockIdx.y;               // b*16 + h2
  const int b    = bh2 >> 4;
  const int hv   = (bh2 & 15) >> 1;          // shared V head

  const __bf16* Qh = qs + (size_t)bh2 * 1024 * 64;
  const __bf16* Kh = ks + (size_t)bh2 * 1024 * 64;
  const __bf16* Vh = vt + ((size_t)(b * 8 + hv)) * 128 * 1024;

  // ---- stage Q stripe [64][64] ----
#pragma unroll
  for (int j = 0; j < 2; ++j) {
    int row = (tid >> 3) + 32 * j;
    int d8  = (tid & 7) * 8;
    *(uint4*)&q_lds[row * QP + d8] = *(const uint4*)&Qh[(size_t)(q0 + row) * 64 + d8];
  }

  // ---- phase 1: scores, 64-key chunks ----
  const int rt    = wave & 3;          // query 16-row tile
  const int cbase = (wave >> 2) * 2;   // first of two key 16-col tiles
  for (int kt = 0; kt < 16; ++kt) {
    __syncthreads();
#pragma unroll
    for (int j = 0; j < 2; ++j) {
      int row = (tid >> 3) + 32 * j;   // 0..63
      int d8  = (tid & 7) * 8;
      *(uint4*)&k_lds[row * KP + d8] =
          *(const uint4*)&Kh[(size_t)(kt * 64 + row) * 64 + d8];
    }
    __syncthreads();

    v16bf afrag[2];
    const __bf16* arow = &q_lds[(rt * 16 + l16) * QP];
#pragma unroll
    for (int kk = 0; kk < 2; ++kk) afrag[kk] = lds_frag_a(arow + kk * 32, half);

#pragma unroll
    for (int ct = 0; ct < 2; ++ct) {
      const __bf16* brow = &k_lds[((cbase + ct) * 16 + l16) * KP];
      v8f acc = v8f_zero();
#pragma unroll
      for (int kk = 0; kk < 2; ++kk)
        acc = WMMA_BF16(afrag[kk], lds_frag_b(brow + kk * 32, half), acc);
      int col = kt * 64 + (cbase + ct) * 16 + l16;
#pragma unroll
      for (int r = 0; r < 8; ++r)
        s_lds[(rt * 16 + r + 8 * half) * SP + col] = (__bf16)acc[r];
    }
  }
  __syncthreads();

  // ---- phase 2: softmax over 1024 keys, one wave per 8 rows ----
  for (int rr = 0; rr < 8; ++rr) {
    int row = wave * 8 + rr;
    __bf16* srow = &s_lds[row * SP];
    float mx = -3.0e30f;
    for (int c = lane; c < 1024; c += 32) mx = fmaxf(mx, (float)srow[c]);
    for (int off = 16; off; off >>= 1) mx = fmaxf(mx, __shfl_xor(mx, off, 32));
    float sum = 0.f;
    for (int c = lane; c < 1024; c += 32) {
      float e = __expf((float)srow[c] - mx);
      sum += e;
      srow[c] = (__bf16)e;
    }
    for (int off = 16; off; off >>= 1) sum += __shfl_xor(sum, off, 32);
    float inv = 1.f / sum;
    for (int c = lane; c < 1024; c += 32)
      srow[c] = (__bf16)((float)srow[c] * inv);
  }
  __syncthreads();

  // ---- phase 3: O = P @ V  (dv = 128), V tiles via TDM ----
  const int r2  = wave & 3;
  const int dh2 = wave >> 2;
  v8f oacc[4];
#pragma unroll
  for (int nt = 0; nt < 4; ++nt) oacc[nt] = v8f_zero();

  for (int mt = 0; mt < 32; ++mt) {
    __syncthreads();
#if USE_TDM
    if (wave == 0) {
      tdm_load_v_tile(Vh + (size_t)mt * 32, (unsigned)V_OFF);
      __builtin_amdgcn_s_wait_tensorcnt((short)0);
    }
#else
#pragma unroll
    for (int j = 0; j < 2; ++j) {
      int row = (tid >> 2) + 64 * j;         // 0..127 (d index)
      int m8  = (tid & 3) * 8;
      *(uint4*)&v_lds[row * VP + m8] =
          *(const uint4*)&Vh[(size_t)row * 1024 + mt * 32 + m8];
    }
#endif
    __syncthreads();

    v16bf a = lds_frag_a(&s_lds[(r2 * 16 + l16) * SP + mt * 32], half);
    v16bf bfrag[4];
#pragma unroll
    for (int nt = 0; nt < 4; ++nt)
      bfrag[nt] = lds_frag_b(&v_lds[(dh2 * 64 + nt * 16 + l16) * VP], half);
#pragma unroll
    for (int nt = 0; nt < 4; ++nt)
      oacc[nt] = WMMA_BF16(a, bfrag[nt], oacc[nt]);
  }

#pragma unroll
  for (int nt = 0; nt < 4; ++nt) {
#pragma unroll
    for (int r = 0; r < 8; ++r) {
      int qr = q0 + r2 * 16 + r + 8 * half;
      int d  = dh2 * 64 + nt * 16 + l16;
      obuf[((size_t)bh2 * 1024 + qr) * 128 + d] = oacc[nt][r];
    }
  }
}

// ---------------------------------------------------------------------------
extern "C" void kernel_launch(void* const* d_in, const int* in_sizes, int n_in,
                              void* d_out, int out_size, void* d_ws, size_t ws_size,
                              hipStream_t stream) {
  (void)in_sizes; (void)n_in; (void)out_size; (void)ws_size;
  const float* query = (const float*)d_in[0];
  const float* key   = (const float*)d_in[1];
  const float* value = (const float*)d_in[2];
  const float* Wq = (const float*)d_in[3];  const float* bq = (const float*)d_in[4];
  const float* Wk = (const float*)d_in[5];  const float* bk = (const float*)d_in[6];
  const float* Wv = (const float*)d_in[7];  const float* bv = (const float*)d_in[8];
  const float* Wo = (const float*)d_in[9];  const float* bo = (const float*)d_in[10];
  const float* lq1 = (const float*)d_in[11];
  const float* lq2 = (const float*)d_in[12];
  const float* lk1 = (const float*)d_in[13];
  const float* lk2 = (const float*)d_in[14];
  // d_in[15] = h (16), hardcoded.

  // ---- workspace carve-up (~57 MB) ----
  char* ws = (char*)d_ws;
  float*  lam  = (float*)ws;
  __bf16* qs   = (__bf16*)(ws + 256);                       // 4*16*1024*64 bf16 = 8 MB
  __bf16* ks   = qs + (size_t)4 * 16 * 1024 * 64;           // 8 MB
  __bf16* vtb  = ks + (size_t)4 * 16 * 1024 * 64;           // 8 MB
  float*  obuf = (float*)((char*)vtb + (size_t)4 * 8 * 128 * 1024 * 2); // 32 MB f32

  lambda_kernel<<<1, 32, 0, stream>>>(lq1, lq2, lk1, lk2, lam);

  dim3 gg(32, 8); // M=4096/128, N=1024/128
  gemm_bf16_kernel<<<gg, 256, 0, stream>>>(query, Wq, bq, lam, 1024, 0, qs, ks, vtb, nullptr);
  gemm_bf16_kernel<<<gg, 256, 0, stream>>>(key,   Wk, bk, lam, 1024, 1, qs, ks, vtb, nullptr);
  gemm_bf16_kernel<<<gg, 256, 0, stream>>>(value, Wv, bv, lam, 1024, 2, qs, ks, vtb, nullptr);

  attn_kernel<<<dim3(16, 64), 256, (size_t)ATTN_LDS, stream>>>(qs, ks, vtb, obuf);

  // final projection with fused differential combine
  gemm_bf16_kernel<<<gg, 256, 0, stream>>>(obuf, Wo, bo, lam, 1024, 3,
                                           nullptr, nullptr, nullptr, (float*)d_out);
}
